// QuantumTransformer_65481071399472
// MI455X (gfx1250) — compile-verified, matrix operand order checked by reading, not run on hardware
//
#include <hip/hip_runtime.h>
#include <hip/hip_bf16.h>
#include <math.h>

typedef __attribute__((ext_vector_type(16))) _Float16 v16h;
typedef __attribute__((ext_vector_type(8)))  _Float16 v8h;
typedef __attribute__((ext_vector_type(8)))  float    v8f;

#define LAYERS   4
#define DMODEL   256
#define HEADS    64
#define NQl      4
#define FFN_DIM  1024
#define NBATCH   16
#define SEQ      2048
#define MTILE    32
#define LN_EPS   1e-5f

// packed f16 weight sizes (elements)
#define PK_COMBINE (LAYERS*DMODEL*DMODEL)    // 262144
#define PK_LIN1    (LAYERS*FFN_DIM*DMODEL)   // 1048576
#define PK_LIN2    (LAYERS*DMODEL*FFN_DIM)   // 1048576
#define PK_TOTAL   (PK_COMBINE+PK_LIN1+PK_LIN2)

// ---------------------------------------------------------------------------
// Weight pre-pack: f32 -> f16, fragment-major layout for v_wmma_f32_16x16x32_f16
// B-tile (16x32 N x K) blob: [lane 0..31][e 0..15] f16 (1 KB per tile).
//   lane<16 : col n = n0+lane,     K offsets {0..7, 16..23}  (e<8 -> e, else e+8)
//   lane>=16: col n = n0+lane-16,  K offsets {8..15, 24..31} (base +8)
// ---------------------------------------------------------------------------
__global__ void pack_weights(const float* __restrict__ cw,
                             const float* __restrict__ w1,
                             const float* __restrict__ w2,
                             _Float16* __restrict__ pk) {
  unsigned idx = blockIdx.x * 256u + threadIdx.x;
  if (idx >= (unsigned)PK_TOTAL) return;
  unsigned l, rem, tile, lane, e, tn, kt, n, k;
  float val;
  if (idx < PK_COMBINE) {                       // [D,D], 16 n-tiles x 8 k-tiles
    l = idx >> 16; rem = idx & 65535u;
    tile = rem >> 9; lane = (rem >> 4) & 31u; e = rem & 15u;
    tn = tile >> 3; kt = tile & 7u;
    n = tn * 16u + (lane & 15u);
    k = kt * 32u + ((lane & 16u) ? 8u : 0u) + ((e < 8u) ? e : e + 8u);
    val = cw[(l * 256u + n) * 256u + k];
  } else if (idx < PK_COMBINE + PK_LIN1) {      // [FFN,D], 64 n-tiles x 8 k-tiles
    unsigned q = idx - PK_COMBINE;
    l = q >> 18; rem = q & 262143u;
    tile = rem >> 9; lane = (rem >> 4) & 31u; e = rem & 15u;
    tn = tile >> 3; kt = tile & 7u;
    n = tn * 16u + (lane & 15u);
    k = kt * 32u + ((lane & 16u) ? 8u : 0u) + ((e < 8u) ? e : e + 8u);
    val = w1[(l * 1024u + n) * 256u + k];
  } else {                                      // [D,FFN], 16 n-tiles x 32 k-tiles
    unsigned q = idx - PK_COMBINE - PK_LIN1;
    l = q >> 18; rem = q & 262143u;
    tile = rem >> 9; lane = (rem >> 4) & 31u; e = rem & 15u;
    tn = tile >> 5; kt = tile & 31u;
    n = tn * 16u + (lane & 15u);
    k = kt * 32u + ((lane & 16u) ? 8u : 0u) + ((e < 8u) ? e : e + 8u);
    val = w2[(l * 256u + n) * 1024u + k];
  }
  pk[idx] = (_Float16)val;
}

__global__ void zero_pooled(float* __restrict__ pooled) {
  pooled[blockIdx.x * 256 + threadIdx.x] = 0.f;
}

// ---------------------------------------------------------------------------
// Fragment loaders
// ---------------------------------------------------------------------------
__device__ __forceinline__ v16h load_a_frag(const _Float16* __restrict__ base,
                                            int m, int k0, int lane) {
  // ISA 16-bit A 16x32: lanes 0-15 K{0..7,16..23}; lanes 16-31 K{8..15,24..31}
  const _Float16* p = base + m * DMODEL + k0 + ((lane & 16) ? 8 : 0);
  v8h lo = *(const v8h*)p;
  v8h hi = *(const v8h*)(p + 16);
  v16h a;
#pragma unroll
  for (int i = 0; i < 8; ++i) { a[i] = lo[i]; a[i + 8] = hi[i]; }
  return a;
}

__device__ __forceinline__ v16h load_b_frag(const _Float16* __restrict__ tile, int lane) {
  const _Float16* p = tile + lane * 16;
  v8h lo = *(const v8h*)p;
  v8h hi = *(const v8h*)(p + 8);
  v16h b;
#pragma unroll
  for (int i = 0; i < 8; ++i) { b[i] = lo[i]; b[i + 8] = hi[i]; }
  return b;
}

__device__ __forceinline__ v8f wmma16(v16h a, v16h b, v8f c) {
  return __builtin_amdgcn_wmma_f32_16x16x32_f16(false, a, false, b, (short)0, c,
                                                false, false);
}

// row-wise LayerNorm over D=256 using 8-lane shuffle groups (wave32)
__device__ __forceinline__ void layernorm_rows(float (*xf)[DMODEL],
                                               _Float16 (*a16)[DMODEL],
                                               const float* __restrict__ sc,
                                               const float* __restrict__ bi,
                                               const float* __restrict__ ffnw,
                                               int write_cos, int tid) {
  int row = tid >> 3, sub = tid & 7;
  float sum = 0.f, sq = 0.f;
#pragma unroll
  for (int j = 0; j < 32; ++j) {
    float v = xf[row][sub + 8 * j];
    sum += v; sq += v * v;
  }
  sum += __shfl_xor(sum, 1, 8); sq += __shfl_xor(sq, 1, 8);
  sum += __shfl_xor(sum, 2, 8); sq += __shfl_xor(sq, 2, 8);
  sum += __shfl_xor(sum, 4, 8); sq += __shfl_xor(sq, 4, 8);
  float mu   = sum * (1.f / DMODEL);
  float var  = sq * (1.f / DMODEL) - mu * mu;
  float rinv = rsqrtf(var + LN_EPS);
#pragma unroll
  for (int j = 0; j < 32; ++j) {
    int d = sub + 8 * j;
    float xn = (xf[row][d] - mu) * rinv * sc[d] + bi[d];
    xf[row][d] = xn;
    if (write_cos) a16[row][d] = (_Float16)(__cosf(ffnw[d]) * __cosf(xn));
  }
}

// ---------------------------------------------------------------------------
// Fused 4-layer network: one workgroup = 32 tokens, all layers in LDS.
// All GEMMs are kt-outer: one A fragment per kt shared by the wave's 4 N-tiles
// (4x less LDS traffic; back-to-back WMMAs reuse the same A operand).
// ---------------------------------------------------------------------------
__global__ __launch_bounds__(256)
void qtrans_main(const int* __restrict__ tokens, const float* __restrict__ emb,
                 const float* __restrict__ attn_w, const float* __restrict__ ffn_w,
                 const float* __restrict__ combine_b,
                 const float* __restrict__ lin1_b, const float* __restrict__ lin2_b,
                 const float* __restrict__ ln1_s, const float* __restrict__ ln1_b,
                 const float* __restrict__ ln2_s, const float* __restrict__ ln2_b,
                 const _Float16* __restrict__ pk_combine,
                 const _Float16* __restrict__ pk_lin1,
                 const _Float16* __restrict__ pk_lin2,
                 float* __restrict__ pooled) {
  __shared__ float    xf[MTILE][DMODEL];     // 32 KB  running activations (f32)
  __shared__ _Float16 a16[MTILE][DMODEL];    // 16 KB  f16 GEMM A input
  __shared__ _Float16 fbuf[MTILE][DMODEL];   // 16 KB  relu'd FFN chunk

  const int tid  = threadIdx.x;
  const int wave = tid >> 5;
  const int lane = tid & 31;
  const int tm   = wave >> 2;         // wave's M-tile (0 or 1), shared by its 4 tiles
  const int tn0  = (wave & 3) * 4;    // wave's first N-tile
  const int mrow = tm * 16 + (lane & 15);
  const int mb   = tm * 16 + ((lane & 16) ? 8 : 0);
  const int ncol = lane & 15;
  const long row0 = (long)blockIdx.x * MTILE;

  // ---- embedding gather + sinusoidal positions ----
  for (int idx = tid; idx < MTILE * DMODEL; idx += 256) {
    int r = idx >> 8, d = idx & 255;
    long g = row0 + r;
    int tok = tokens[g];
    int s = (int)(g & (SEQ - 1));
    float freq = __expf(-0.03597789156f * (float)(d & ~1));  // -ln(10000)/256
    float ang  = (float)s * freq;
    float pe   = (d & 1) ? __cosf(ang) : __sinf(ang);
    xf[r][d] = emb[(long)tok * DMODEL + d] + pe;
  }
  __syncthreads();

  for (int l = 0; l < LAYERS; ++l) {
    // ---- quantum attention features: cumprod of cos over NQ=4 per head ----
    const float* aw = attn_w + l * HEADS * NQl;
    for (int idx = tid; idx < MTILE * HEADS; idx += 256) {
      int r = idx >> 6, h = idx & 63;
      float p = 1.f;
#pragma unroll
      for (int j = 0; j < NQl; ++j) {
        p *= __cosf(aw[h * NQl + j]) * __cosf(xf[r][h * NQl + j]);
        a16[r][h * NQl + j] = (_Float16)p;
      }
    }
    __syncthreads();

    // ---- GEMM1: a @ combine_w^T, residual into xf ----
    const _Float16* pkc = pk_combine + (long)l * DMODEL * DMODEL;
    const float* cb = combine_b + l * DMODEL;
    {
      v8f acc[4];
#pragma unroll
      for (int j = 0; j < 4; ++j)
#pragma unroll
        for (int i = 0; i < 8; ++i) acc[j][i] = 0.f;
#pragma unroll
      for (int kt = 0; kt < 8; ++kt) {
        v16h A = load_a_frag(&a16[0][0], mrow, kt * 32, lane);
#pragma unroll
        for (int j = 0; j < 4; ++j) {
          v16h Bf = load_b_frag(pkc + ((((tn0 + j) << 3) + kt) << 9), lane);
          acc[j] = wmma16(A, Bf, acc[j]);
        }
      }
#pragma unroll
      for (int j = 0; j < 4; ++j) {
        int n = (tn0 + j) * 16 + ncol;
        float bias = cb[n];
#pragma unroll
        for (int i = 0; i < 8; ++i) xf[mb + i][n] += acc[j][i] + bias;
      }
    }
    __syncthreads();
    layernorm_rows(xf, a16, ln1_s + l * DMODEL, ln1_b + l * DMODEL,
                   ffn_w + l * DMODEL, 1, tid);
    __syncthreads();

    // ---- fused FFN: 4 chunks of 256; lin2 accumulators persist in VGPRs ----
    const _Float16* pk1 = pk_lin1 + (long)l * FFN_DIM * DMODEL;
    const _Float16* pk2 = pk_lin2 + (long)l * DMODEL * FFN_DIM;
    const float* b1 = lin1_b + l * FFN_DIM;
    const float* b2 = lin2_b + l * DMODEL;
    v8f acc3[4];
#pragma unroll
    for (int j = 0; j < 4; ++j) {
      float bias2 = b2[(tn0 + j) * 16 + ncol];
#pragma unroll
      for (int i = 0; i < 8; ++i) acc3[j][i] = bias2;
    }
    for (int c = 0; c < 4; ++c) {
      // GEMM2 chunk: relu(a16 @ lin1_c^T + b1) -> fbuf
      {
        v8f facc[4];
#pragma unroll
        for (int j = 0; j < 4; ++j)
#pragma unroll
          for (int i = 0; i < 8; ++i) facc[j][i] = 0.f;
#pragma unroll
        for (int kt = 0; kt < 8; ++kt) {
          if (kt < 7)
            __builtin_prefetch(
                pk1 + ((((c * 16 + tn0) << 3) + kt + 1) << 9) + lane * 16, 0, 0);
          v16h A = load_a_frag(&a16[0][0], mrow, kt * 32, lane);
#pragma unroll
          for (int j = 0; j < 4; ++j) {
            int gn = c * 16 + tn0 + j;
            v16h Bf = load_b_frag(pk1 + (((gn << 3) + kt) << 9), lane);
            facc[j] = wmma16(A, Bf, facc[j]);
          }
        }
#pragma unroll
        for (int j = 0; j < 4; ++j) {
          int gn = c * 16 + tn0 + j;
          float bias = b1[gn * 16 + ncol];
          int nn = (tn0 + j) * 16 + ncol;
#pragma unroll
          for (int i = 0; i < 8; ++i) {
            float v = facc[j][i] + bias;
            fbuf[mb + i][nn] = (_Float16)(v > 0.f ? v : 0.f);
          }
        }
      }
      __syncthreads();
      // GEMM3 accumulate: y += fbuf @ lin2_c^T
#pragma unroll
      for (int kt = 0; kt < 8; ++kt) {
        int kk = c * 8 + kt;
        v16h A = load_a_frag(&fbuf[0][0], mrow, kt * 32, lane);
#pragma unroll
        for (int j = 0; j < 4; ++j) {
          v16h Bf = load_b_frag(pk2 + ((((tn0 + j) << 5) + kk) << 9), lane);
          acc3[j] = wmma16(A, Bf, acc3[j]);
        }
      }
      __syncthreads();
    }
    // residual + LN2
#pragma unroll
    for (int j = 0; j < 4; ++j) {
      int n = (tn0 + j) * 16 + ncol;
#pragma unroll
      for (int i = 0; i < 8; ++i) xf[mb + i][n] += acc3[j][i];
    }
    __syncthreads();
    layernorm_rows(xf, a16, ln2_s + l * DMODEL, ln2_b + l * DMODEL,
                   ffn_w + l * DMODEL, 0, tid);
    __syncthreads();
  }

  // ---- mean-pool contribution (whole tile is inside one batch element) ----
  {
    int b = (int)(row0 / SEQ);
    float s = 0.f;
#pragma unroll
    for (int r = 0; r < MTILE; ++r) s += xf[r][tid];
    atomicAdd(&pooled[b * DMODEL + tid], s * (1.f / SEQ));
  }
}

// ---------------------------------------------------------------------------
// pooled[b,:] . cls_w + cls_b -> out[b]
// ---------------------------------------------------------------------------
__global__ void cls_kernel(const float* __restrict__ pooled,
                           const float* __restrict__ cls_w,
                           const float* __restrict__ cls_b,
                           float* __restrict__ out) {
  __shared__ float red[256];
  int b = blockIdx.x, tid = threadIdx.x;
  red[tid] = pooled[b * DMODEL + tid] * cls_w[tid];
  __syncthreads();
  for (int s = 128; s > 0; s >>= 1) {
    if (tid < s) red[tid] += red[tid + s];
    __syncthreads();
  }
  if (tid == 0) out[b] = red[0] + cls_b[0];
}

extern "C" void kernel_launch(void* const* d_in, const int* in_sizes, int n_in,
                              void* d_out, int out_size, void* d_ws, size_t ws_size,
                              hipStream_t stream) {
  const int*   tokens    = (const int*)  d_in[0];
  const float* emb       = (const float*)d_in[1];
  const float* attn_w    = (const float*)d_in[2];
  const float* ffn_w     = (const float*)d_in[3];
  const float* combine_w = (const float*)d_in[4];
  const float* combine_b = (const float*)d_in[5];
  const float* lin1_w    = (const float*)d_in[6];
  const float* lin1_b    = (const float*)d_in[7];
  const float* lin2_w    = (const float*)d_in[8];
  const float* lin2_b    = (const float*)d_in[9];
  const float* ln1_s     = (const float*)d_in[10];
  const float* ln1_bv    = (const float*)d_in[11];
  const float* ln2_s     = (const float*)d_in[12];
  const float* ln2_bv    = (const float*)d_in[13];
  const float* cls_w     = (const float*)d_in[14];
  const float* cls_b     = (const float*)d_in[15];
  float* out = (float*)d_out;

  _Float16* pk         = (_Float16*)d_ws;
  _Float16* pk_combine = pk;
  _Float16* pk_lin1    = pk + PK_COMBINE;
  _Float16* pk_lin2    = pk + PK_COMBINE + PK_LIN1;
  float*    pooled     = (float*)(pk + PK_TOTAL);   // 16*256 f32

  pack_weights<<<(PK_TOTAL + 255) / 256, 256, 0, stream>>>(combine_w, lin1_w,
                                                           lin2_w, pk);
  zero_pooled<<<NBATCH, 256, 0, stream>>>(pooled);
  qtrans_main<<<(NBATCH * SEQ) / MTILE, 256, 0, stream>>>(
      tokens, emb, attn_w, ffn_w, combine_b, lin1_b, lin2_b,
      ln1_s, ln1_bv, ln2_s, ln2_bv, pk_combine, pk_lin1, pk_lin2, pooled);
  cls_kernel<<<NBATCH, 256, 0, stream>>>(pooled, cls_w, cls_b, out);
}
